// VQ_Payam_EMA_8821862826424
// MI455X (gfx1250) — matrix operand order repaired and checked by reading, not compiled, not run on hardware
//
#include <hip/hip_runtime.h>
#include <hip/hip_bf16.h>
#include <stdint.h>

// ---------------------------------------------------------------------------
// VQ-VAE quantization for MI455X (gfx1250, wave32).
//   inputs      : [32768, 512]  f32
//   emb_weight  : [2048, 512]   f32
//   outputs (flat, in order): loss(1), quantized_st(32768*512),
//                             perplexity(1), encodings(32768*2048)
// Strategy: bf16 WMMA 16x16x32 GEMM (X . E^T) fused with streaming argmax of
// (x.e - 0.5*||e||^2); E tiles double-buffered in LDS via async global->LDS.
// Workspace use: ~36 MB (X bf16 32MB, E bf16 2MB, norms/indices/counts).
// ---------------------------------------------------------------------------

#define NTOK 32768
#define DIM  512
#define NCODE 2048
#define ROWS_PER_WG 128          // 8 waves x 16 rows
#define CHUNK 16                 // codes per LDS chunk
#define NCHUNK (NCODE / CHUNK)   // 128
#define CHUNK_BYTES (CHUNK * DIM * 2)      // 16 KB bf16
#define CHUNK_GRAN  (CHUNK_BYTES / 16)     // 1024 x b128

typedef __attribute__((ext_vector_type(16))) __bf16 v16bf;
typedef __attribute__((ext_vector_type(8)))  float  v8f;
typedef __attribute__((ext_vector_type(4)))  unsigned int u32x4;

union Frag { u32x4 u[2]; v16bf v; };

__device__ __forceinline__ unsigned short f2bf(float f) {
  unsigned u = __float_as_uint(f);
  unsigned r = u + 0x7FFFu + ((u >> 16) & 1u);   // round-to-nearest-even
  return (unsigned short)(r >> 16);
}

// ---------------------------------------------------------------------------
// Prep: f32 -> bf16 packer (8 floats -> one 16B granule)
// ---------------------------------------------------------------------------
__global__ __launch_bounds__(256) void vq_pack_bf16(
    const float4* __restrict__ src, u32x4* __restrict__ dst, int nGran) {
  int i = blockIdx.x * 256 + threadIdx.x;
  if (i >= nGran) return;
  float4 a = src[2 * i], b = src[2 * i + 1];
  u32x4 o;
  o.x = (unsigned)f2bf(a.x) | ((unsigned)f2bf(a.y) << 16);
  o.y = (unsigned)f2bf(a.z) | ((unsigned)f2bf(a.w) << 16);
  o.z = (unsigned)f2bf(b.x) | ((unsigned)f2bf(b.y) << 16);
  o.w = (unsigned)f2bf(b.z) | ((unsigned)f2bf(b.w) << 16);
  dst[i] = o;
}

// 0.5 * ||e_k||^2 per code row
__global__ __launch_bounds__(256) void vq_enorm(
    const float* __restrict__ E, float* __restrict__ hn) {
  int r = blockIdx.x, tid = threadIdx.x;
  const float* er = E + (size_t)r * DIM;
  float s = 0.f;
  for (int d = tid; d < DIM; d += 256) { float v = er[d]; s += v * v; }
  for (int m = 16; m >= 1; m >>= 1) s += __shfl_xor(s, m, 32);
  __shared__ float wsum[8];
  if ((tid & 31) == 0) wsum[tid >> 5] = s;
  __syncthreads();
  if (tid == 0) {
    float t = 0.f;
    for (int w = 0; w < 8; ++w) t += wsum[w];
    hn[r] = 0.5f * t;
  }
}

__global__ void vq_zero_accum(float* __restrict__ counts, float* __restrict__ sumsq) {
  int t = blockIdx.x * 256 + threadIdx.x;
  if (t < NCODE) counts[t] = 0.f;
  if (t == 0) *sumsq = 0.f;
}

// ---------------------------------------------------------------------------
// Main: bf16 WMMA GEMM fused with argmax(x.e - 0.5||e||^2)
// ---------------------------------------------------------------------------
__device__ __forceinline__ void async_copy_chunk(
    const unsigned char* __restrict__ ebase, int chunk, void* ldsdst, int tid) {
  unsigned lds0 = (unsigned)(size_t)ldsdst;       // low 32 bits == LDS offset
  uint64_t g0 = (uint64_t)(size_t)ebase + (uint64_t)chunk * CHUNK_BYTES;
#pragma unroll
  for (int i = 0; i < CHUNK_GRAN / 256; ++i) {    // 4 x b128 per thread
    unsigned g = (unsigned)tid + i * 256u;
    unsigned lds = lds0 + g * 16u;
    uint64_t ga = g0 + (uint64_t)g * 16u;
    asm volatile("global_load_async_to_lds_b128 %0, %1, off"
                 :: "v"(lds), "v"(ga) : "memory");
  }
}

__global__ __launch_bounds__(256) void vq_gemm_argmax(
    const u32x4* __restrict__ Xb,                 // bf16 X, 16B granules
    const unsigned char* __restrict__ Eb,         // bf16 E, row-major bytes
    const float* __restrict__ half_norm,          // [NCODE]
    int* __restrict__ outIdx)                     // [NTOK]
{
  __shared__ u32x4 ebuf[2][CHUNK_GRAN];           // 2 x 16KB double buffer
  const int tid  = threadIdx.x;
  const int wave = tid >> 5;
  const int lane = tid & 31;
  const int half = lane >> 4;                     // 0: lanes 0-15, 1: 16-31
  const int l15  = lane & 15;
  const int wrow = blockIdx.x * ROWS_PER_WG + wave * 16;

  // ---- preload A fragments (16 rows x 512 K) per documented 16-bit A layout:
  // lane l (M = l&15) holds K = [32s+off, +8) and [32s+16+off, +8), off = 8*half
  Frag A[16];
  {
    const size_t rbase = (size_t)(wrow + l15) * DIM;
    const int off = half * 8;
#pragma unroll
    for (int s = 0; s < 16; ++s) {
      A[s].u[0] = Xb[(rbase + 32 * s + off) >> 3];
      A[s].u[1] = Xb[(rbase + 32 * s + 16 + off) >> 3];
    }
  }

  float best[8];
  int   bidx[8];
#pragma unroll
  for (int r = 0; r < 8; ++r) { best[r] = -3.4e38f; bidx[r] = 0; }

  async_copy_chunk(Eb, 0, &ebuf[0][0], tid);

  for (int c = 0; c < NCHUNK; ++c) {
    const int buf = c & 1;
    asm volatile("s_wait_asynccnt 0x0" ::: "memory");
    __syncthreads();                               // chunk c resident in LDS
    if (c + 1 < NCHUNK)
      async_copy_chunk(Eb, c + 1, &ebuf[buf ^ 1][0], tid);

    // B layout (32x16 bf16): lane holds N = l&15, K = [32s+8*half, +8) and
    // [32s+16+8*half, +8) of row (l&15) of the E chunk.
    const u32x4* __restrict__ eb = &ebuf[buf][0];
    const int gb = l15 * 64 + half;                // granule base for this lane
    v8f acc0 = {0.f, 0.f, 0.f, 0.f, 0.f, 0.f, 0.f, 0.f};
    v8f acc1 = {0.f, 0.f, 0.f, 0.f, 0.f, 0.f, 0.f, 0.f};
#pragma unroll
    for (int s = 0; s < 16; s += 2) {
      Frag B0, B1;
      int g0 = gb + 4 * s;
      B0.u[0] = eb[g0];     B0.u[1] = eb[g0 + 2];
      int g1 = g0 + 4;
      B1.u[0] = eb[g1];     B1.u[1] = eb[g1 + 2];
      acc0 = __builtin_amdgcn_wmma_f32_16x16x32_bf16(
          false, A[s].v,     false, B0.v, (short)0, acc0, false, false);
      acc1 = __builtin_amdgcn_wmma_f32_16x16x32_bf16(
          false, A[s + 1].v, false, B1.v, (short)0, acc1, false, false);
    }

    // C/D layout: VGPR r -> M = r + 8*half, N = l&15.
    const int n = c * CHUNK + l15;
    const float hn = half_norm[n];
#pragma unroll
    for (int r = 0; r < 8; ++r) {
      float sc = acc0[r] + acc1[r] - hn;
      if (sc > best[r]) { best[r] = sc; bidx[r] = n; }   // first-min tiebreak
    }
  }

  // reduce across the 16 lanes of each half (same M, different N)
#pragma unroll
  for (int r = 0; r < 8; ++r) {
    float s = best[r];
    int   i = bidx[r];
#pragma unroll
    for (int m = 8; m >= 1; m >>= 1) {
      float s2 = __shfl_xor(s, m, 32);
      int   i2 = __shfl_xor(i, m, 32);
      if (s2 > s || (s2 == s && i2 < i)) { s = s2; i = i2; }
    }
    best[r] = s; bidx[r] = i;
  }
  if (l15 == 0) {
#pragma unroll
    for (int r = 0; r < 8; ++r) outIdx[wrow + half * 8 + r] = bidx[r];
  }
}

// ---------------------------------------------------------------------------
// Epilogue: gather quantized rows, one-hot encodings, loss partials, counts
// ---------------------------------------------------------------------------
__global__ __launch_bounds__(256) void vq_finalize_rows(
    const float* __restrict__ X, const float* __restrict__ E,
    const int* __restrict__ idx, float* __restrict__ outQ,
    float* __restrict__ outEnc, float* __restrict__ counts,
    float* __restrict__ sumsq) {
  const int n = blockIdx.x, tid = threadIdx.x;
  const int k = idx[n];
  const float* __restrict__ er = E + (size_t)k * DIM;
  const float* __restrict__ xr = X + (size_t)n * DIM;
  float* __restrict__ qr = outQ + (size_t)n * DIM;
  float local = 0.f;
  for (int d = tid; d < DIM; d += 256) {
    float q = er[d];
    float diff = q - xr[d];
    qr[d] = q;                 // straight-through value == quantized
    local += diff * diff;
  }
  float* __restrict__ encr = outEnc + (size_t)n * NCODE;
  for (int j = tid; j < NCODE; j += 256) encr[j] = (j == k) ? 1.0f : 0.0f;

  for (int m = 16; m >= 1; m >>= 1) local += __shfl_xor(local, m, 32);
  __shared__ float wsum[8];
  if ((tid & 31) == 0) wsum[tid >> 5] = local;
  __syncthreads();
  if (tid == 0) {
    float t = 0.f;
    for (int w = 0; w < 8; ++w) t += wsum[w];
    atomicAdd(sumsq, t);
    atomicAdd(&counts[k], 1.0f);
  }
}

__global__ __launch_bounds__(256) void vq_scalars(
    const float* __restrict__ counts, const float* __restrict__ sumsq,
    float* __restrict__ out_loss, float* __restrict__ out_ppl) {
  const int tid = threadIdx.x;
  float h = 0.f;
  for (int j = tid; j < NCODE; j += 256) {
    float p = counts[j] * (1.0f / (float)NTOK);
    h += p * logf(p + 1e-10f);
  }
  for (int m = 16; m >= 1; m >>= 1) h += __shfl_xor(h, m, 32);
  __shared__ float wsum[8];
  if ((tid & 31) == 0) wsum[tid >> 5] = h;
  __syncthreads();
  if (tid == 0) {
    float t = 0.f;
    for (int w = 0; w < 8; ++w) t += wsum[w];
    *out_ppl  = expf(-t);
    *out_loss = 0.25f * (*sumsq) / (float)((size_t)NTOK * DIM);
  }
}

// ---------------------------------------------------------------------------
extern "C" void kernel_launch(void* const* d_in, const int* in_sizes, int n_in,
                              void* d_out, int out_size, void* d_ws, size_t ws_size,
                              hipStream_t stream) {
  const float* X = (const float*)d_in[0];     // [32768, 512]
  const float* E = (const float*)d_in[1];     // [2048, 512]
  float* out = (float*)d_out;
  float* out_loss = out;                                  // [1]
  float* out_q    = out + 1;                              // [32768*512]
  float* out_ppl  = out + 1 + (size_t)NTOK * DIM;         // [1]
  float* out_enc  = out + 2 + (size_t)NTOK * DIM;         // [32768*2048]

  // workspace layout (~35.8 MB)
  char* w = (char*)d_ws;
  u32x4*         Xb     = (u32x4*)w;                             // 32 MB bf16 X
  unsigned char* Eb     = (unsigned char*)(w + 33554432);        // 2 MB bf16 E
  float*         hn     = (float*)(w + 35651584);                // 8 KB
  int*           idx    = (int*)(w + 35659776);                  // 128 KB
  float*         counts = (float*)(w + 35790848);                // 8 KB
  float*         sumsq  = (float*)(w + 35799040);                // 4 B

  // 1) zero accumulators (ws is poisoned between runs)
  vq_zero_accum<<<(NCODE + 255) / 256, 256, 0, stream>>>(counts, sumsq);
  // 2) f32 -> bf16 for X and E
  const int xGran = NTOK * DIM / 8;   // 2,097,152
  const int eGran = NCODE * DIM / 8;  // 131,072
  vq_pack_bf16<<<xGran / 256, 256, 0, stream>>>((const float4*)X, Xb, xGran);
  vq_pack_bf16<<<eGran / 256, 256, 0, stream>>>((const float4*)E, (u32x4*)Eb, eGran);
  // 3) 0.5 * ||e||^2
  vq_enorm<<<NCODE, 256, 0, stream>>>(E, hn);
  // 4) fused WMMA GEMM + argmax -> nearest code per token
  vq_gemm_argmax<<<NTOK / ROWS_PER_WG, 256, 0, stream>>>(Xb, Eb, hn, idx);
  // 5) gather + one-hot + loss partials + counts
  vq_finalize_rows<<<NTOK, 256, 0, stream>>>(X, E, idx, out_q, out_enc,
                                             counts, sumsq);
  // 6) loss + perplexity
  vq_scalars<<<1, 256, 0, stream>>>(counts, sumsq, out_loss, out_ppl);
}
